// PlanFormerDecoderLayer_18992345383476
// MI455X (gfx1250) — matrix operand order repaired and checked by reading, not compiled, not run on hardware
//
#include <hip/hip_runtime.h>
#include <math.h>

// ---------------------------------------------------------------------------
// PlanFormer decoder layer for MI455X (gfx1250), wave32 + WMMA f16.
// GEMMs: 128x64 block tile, 8 waves x (32x32) strips, BK=32,
//        v_wmma_f32_16x16x32_f16, async global->LDS A staging when available.
// ---------------------------------------------------------------------------

typedef __attribute__((ext_vector_type(16))) _Float16 v16h;
typedef __attribute__((ext_vector_type(8)))  float    v8f;

#define NB    4
#define QN    1024
#define EE    256
#define NHH   8
#define HDD   32
#define PPTS  4
#define HBEV  200
#define WBEV  200
#define HWB   (HBEV * WBEV)
#define DFFN  512
#define MROWS (NB * QN)          // 4096 query rows

#if __has_builtin(__builtin_amdgcn_global_load_async_to_lds_b128) && \
    __has_builtin(__builtin_amdgcn_s_wait_asynccnt)
#define USE_ASYNC_LDS 1
// b128 variant moves one int4 per lane; params are (global int4*, lds int4*,
// imm offset, imm cpol) per the toolchain diagnostic.
typedef int v4i __attribute__((vector_size(16)));
typedef __attribute__((address_space(1))) v4i gv4i;
typedef __attribute__((address_space(3))) v4i lv4i;
#else
#define USE_ASYNC_LDS 0
#endif

__device__ __forceinline__ v8f wmma16(v16h a, v16h b, v8f c) {
  // D = A(16x32 f16) x B(32x16 f16) + C(16x16 f32)
  return __builtin_amdgcn_wmma_f32_16x16x32_f16(false, a, false, b,
                                                (short)0, c, false, false);
}

// ---------------------------------------------------------------------------
// WMMA GEMM: C[M,Nc] = A[M,K] @ B[K,Nc] + bias, optional ReLU, optional f16
// output. Block tile 128x64, BK=32, 256 threads = 8 waves, each wave 32x32.
// Requires M % 128 == 0, K % 32 == 0 (true for all calls); Nc guarded.
// A tile staged raw f32 via async global->LDS (converted during fragment
// build); B tile staged synchronously with f32->f16 convert + transpose.
// ---------------------------------------------------------------------------
__global__ __launch_bounds__(256)
void gemm_wmma_kernel(const float* __restrict__ A, const float* __restrict__ B,
                      const float* __restrict__ bias,
                      float* __restrict__ Cf, _Float16* __restrict__ Ch,
                      int M, int Nc, int K, int relu) {
  __shared__ float    As[128 * 32];  // raw f32 [row][k]   (16 KB)
  __shared__ _Float16 Bs[64 * 32];   // transposed [n][k]  (4 KB)

  const int tid  = threadIdx.x;
  const int lane = tid & 31;
  const int w    = tid >> 5;          // wave 0..7
  const int wm   = w >> 1;            // 0..3  (32-row strip)
  const int wn   = w & 1;             // 0..1  (32-col strip)
  const int bm   = blockIdx.y * 128;
  const int bn   = blockIdx.x * 64;
  const int col  = lane & 15;
  const int hi   = lane >> 4;         // lane half (0/1)

  v8f c00 = {}, c01 = {}, c10 = {}, c11 = {};

  for (int k0 = 0; k0 < K; k0 += 32) {
    // ---- stage A tile (128x32 f32, raw copy), 1024 float4 chunks ----
#pragma unroll
    for (int it = 0; it < 4; ++it) {
      int chunk = tid + it * 256;                 // 0..1023
      int r  = chunk >> 3;                        // row in tile (0..127)
      int kc = (chunk & 7) * 4;                   // k offset
      const float* gp = A + (size_t)(bm + r) * K + k0 + kc;
      float* lp = &As[r * 32 + kc];
#if USE_ASYNC_LDS
      __builtin_amdgcn_global_load_async_to_lds_b128(
          (gv4i*)gp, (lv4i*)lp, 0, 0);
#else
      *(float4*)lp = *(const float4*)gp;
#endif
    }

    // ---- stage B tile (32x64 f32 -> f16, transposed into Bs) ----
#pragma unroll
    for (int it = 0; it < 2; ++it) {
      int chunk = tid + it * 256;                 // 0..511
      int kr = chunk >> 4;                        // k row 0..31
      int nc = (chunk & 15) * 4;                  // col offset 0..60
      int gn = bn + nc;
      float fx = 0.f, fy = 0.f, fz = 0.f, fw = 0.f;
      const float* brow = B + (size_t)(k0 + kr) * Nc;
      if (gn + 3 < Nc) {
        float4 f = *(const float4*)(brow + gn);
        fx = f.x; fy = f.y; fz = f.z; fw = f.w;
      } else {
        if (gn + 0 < Nc) fx = brow[gn + 0];
        if (gn + 1 < Nc) fy = brow[gn + 1];
        if (gn + 2 < Nc) fz = brow[gn + 2];
        if (gn + 3 < Nc) fw = brow[gn + 3];
      }
      Bs[(nc + 0) * 32 + kr] = (_Float16)fx;
      Bs[(nc + 1) * 32 + kr] = (_Float16)fy;
      Bs[(nc + 2) * 32 + kr] = (_Float16)fz;
      Bs[(nc + 3) * 32 + kr] = (_Float16)fw;
    }
    // prefetch next B tile rows into cache (global_prefetch_b8)
    if (k0 + 32 < K)
      __builtin_prefetch(B + (size_t)(k0 + 32 + (tid & 31)) * Nc + bn, 0, 1);

#if USE_ASYNC_LDS
    __builtin_amdgcn_s_wait_asynccnt(0);
#endif
    __syncthreads();

    // ---- build fragments (ISA 7.12.2 layouts) ----
    const int kA = hi ? 8 : 0;
    v16h a0, a1, b0, b1;
    {
      const float* ap0 = &As[(wm * 32 + col) * 32 + kA];
      const float* ap1 = &As[(wm * 32 + 16 + col) * 32 + kA];
#pragma unroll
      for (int i = 0; i < 8; ++i) {
        a0[i]     = (_Float16)ap0[i];
        a0[8 + i] = (_Float16)ap0[16 + i];
        a1[i]     = (_Float16)ap1[i];
        a1[8 + i] = (_Float16)ap1[16 + i];
      }
    }
    {
      const _Float16* bp0 = &Bs[(wn * 32 + col) * 32 + hi * 8];
      const _Float16* bp1 = &Bs[(wn * 32 + 16 + col) * 32 + hi * 8];
#pragma unroll
      for (int i = 0; i < 8; ++i) {
        b0[i] = bp0[i]; b0[8 + i] = bp0[16 + i];
        b1[i] = bp1[i]; b1[8 + i] = bp1[16 + i];
      }
    }
    c00 = wmma16(a0, b0, c00);
    c01 = wmma16(a0, b1, c01);
    c10 = wmma16(a1, b0, c10);
    c11 = wmma16(a1, b1, c11);
    __syncthreads();
  }

  // ---- epilogue: bias (+relu), f32 and/or f16 stores ----
  const int cc0 = bn + wn * 32 + col;
  const int cc1 = cc0 + 16;
#pragma unroll
  for (int sub = 0; sub < 2; ++sub) {
    const int rowb = bm + wm * 32 + sub * 16 + hi * 8;
    const v8f& d0 = sub ? c10 : c00;
    const v8f& d1 = sub ? c11 : c01;
#pragma unroll
    for (int r = 0; r < 8; ++r) {
      int row = rowb + r;
      if (cc0 < Nc) {
        float v = d0[r] + bias[cc0];
        if (relu) v = fmaxf(v, 0.f);
        if (Cf) Cf[(size_t)row * Nc + cc0] = v;
        if (Ch) Ch[(size_t)row * Nc + cc0] = (_Float16)v;
      }
      if (cc1 < Nc) {
        float v = d1[r] + bias[cc1];
        if (relu) v = fmaxf(v, 0.f);
        if (Cf) Cf[(size_t)row * Nc + cc1] = v;
        if (Ch) Ch[(size_t)row * Nc + cc1] = (_Float16)v;
      }
    }
  }
}

// ---------------------------------------------------------------------------
// Flash self-attention: one wave per 16-query tile, per (batch, head).
// HD=32 == WMMA K, so QK^T is one wmma per 16-key tile; PV streams 32 keys
// per wmma with V pre-transposed to (b,h,hd,q) f16.
// ---------------------------------------------------------------------------
__global__ __launch_bounds__(128)
void flash_attn_kernel(const _Float16* __restrict__ qh,
                       const _Float16* __restrict__ kh,
                       const _Float16* __restrict__ vt,
                       float* __restrict__ ctx) {
  __shared__ _Float16 Pt[4][16 * 32];           // per-wave P tile [row][key]

  const int lane = threadIdx.x & 31;
  const int w    = threadIdx.x >> 5;
  const int qt   = blockIdx.x * 4 + w;          // 0..63
  const int h    = blockIdx.y;
  const int b    = blockIdx.z;
  const int col  = lane & 15;
  const int hi   = lane >> 4;

  // Q fragment (A layout): row = qt*16+col, K dim = head-dim
  v16h qa;
  {
    const int kA = hi ? 8 : 0;
    const _Float16* base =
        qh + (size_t)(b * QN + qt * 16 + col) * EE + h * HDD;
#pragma unroll
    for (int i = 0; i < 8; ++i) { qa[i] = base[kA + i]; qa[8 + i] = base[kA + 16 + i]; }
  }

  float m_i[8], l_i[8];
#pragma unroll
  for (int r = 0; r < 8; ++r) { m_i[r] = -1e30f; l_i[r] = 0.f; }
  v8f o0 = {}; v8f o1 = {};
  const float scale = 0.17677669529663687f;     // 1/sqrt(32)

  for (int kc = 0; kc < QN; kc += 32) {
    // K^T fragments (B layout): column = key, K dim = head-dim (table order)
    v16h kf0, kf1;
    {
      const _Float16* kb0 =
          kh + (size_t)(b * QN + kc + col) * EE + h * HDD + hi * 8;
      const _Float16* kb1 = kb0 + (size_t)16 * EE;
#pragma unroll
      for (int i = 0; i < 8; ++i) {
        kf0[i] = kb0[i]; kf0[8 + i] = kb0[16 + i];
        kf1[i] = kb1[i]; kf1[8 + i] = kb1[16 + i];
      }
    }
    v8f z = {};
    v8f s0 = wmma16(qa, kf0, z);
    v8f s1 = wmma16(qa, kf1, z);

    // online softmax per row (rows live on 16-lane halves; xor<=8 stays inside)
#pragma unroll
    for (int r = 0; r < 8; ++r) {
      float a0 = s0[r] * scale, a1 = s1[r] * scale;
      float mx = fmaxf(a0, a1);
#pragma unroll
      for (int msk = 1; msk <= 8; msk <<= 1) mx = fmaxf(mx, __shfl_xor(mx, msk));
      float nm = fmaxf(m_i[r], mx);
      float p0 = __expf(a0 - nm), p1 = __expf(a1 - nm);
      float rs = p0 + p1;
#pragma unroll
      for (int msk = 1; msk <= 8; msk <<= 1) rs += __shfl_xor(rs, msk);
      float corr = __expf(m_i[r] - nm);
      l_i[r] = l_i[r] * corr + rs;
      o0[r] *= corr; o1[r] *= corr;
      m_i[r] = nm;
      int row = r + hi * 8;
      Pt[w][row * 32 + col]      = (_Float16)p0;
      Pt[w][row * 32 + 16 + col] = (_Float16)p1;
    }

    // P as A fragment (K dim = 32 keys), from LDS (same-wave DS ordering)
    v16h pa;
    {
      const int kA = hi ? 8 : 0;
      const _Float16* pb = &Pt[w][col * 32 + kA];
#pragma unroll
      for (int i = 0; i < 8; ++i) { pa[i] = pb[i]; pa[8 + i] = pb[16 + i]; }
    }
    // V fragments from transposed V (b,h,hd,q): keys in table order per lane
    v16h vf0, vf1;
    {
      const _Float16* vb0 =
          vt + ((size_t)((b * NHH + h) * HDD) + col) * QN + kc + hi * 8;
      const _Float16* vb1 = vb0 + (size_t)16 * QN;
#pragma unroll
      for (int i = 0; i < 8; ++i) {
        vf0[i] = vb0[i]; vf0[8 + i] = vb0[16 + i];
        vf1[i] = vb1[i]; vf1[8 + i] = vb1[16 + i];
      }
    }
    o0 = wmma16(pa, vf0, o0);
    o1 = wmma16(pa, vf1, o1);
  }

  // normalize and write ctx (b,q,h,hd) f32
#pragma unroll
  for (int r = 0; r < 8; ++r) {
    int row = b * QN + qt * 16 + r + hi * 8;
    float inv = 1.f / l_i[r];
    ctx[(size_t)row * EE + h * HDD + col]      = o0[r] * inv;
    ctx[(size_t)row * EE + h * HDD + 16 + col] = o1[r] * inv;
  }
}

// ---------------------------------------------------------------------------
// Elementwise helpers
// ---------------------------------------------------------------------------
// v (b,q,h,hd) f32 -> vt (b,h,hd,q) f16
__global__ void transpose_v_kernel(const float* __restrict__ v,
                                   _Float16* __restrict__ vt) {
  int i = blockIdx.x * blockDim.x + threadIdx.x;   // over N*NH*HD*Q
  int q  = i & (QN - 1);
  int t  = i >> 10;
  int hd = t & (HDD - 1);
  t >>= 5;
  int h  = t & (NHH - 1);
  int b  = t >> 3;
  vt[i] = (_Float16)v[(size_t)(b * QN + q) * EE + h * HDD + hd];
}

// out = LayerNorm(a + b) * g + beta ; one block (E=256 threads) per row
__global__ __launch_bounds__(256)
void ln_residual_kernel(const float* __restrict__ a, const float* __restrict__ bsrc,
                        const float* __restrict__ g, const float* __restrict__ beta,
                        float* __restrict__ out) {
  __shared__ float red[256];
  const int row = blockIdx.x, tid = threadIdx.x;
  const size_t base = (size_t)row * EE;
  float x = a[base + tid] + bsrc[base + tid];

  red[tid] = x; __syncthreads();
  for (int s = 128; s > 0; s >>= 1) {
    if (tid < s) red[tid] += red[tid + s];
    __syncthreads();
  }
  float mean = red[0] * (1.f / EE);
  __syncthreads();

  float d = x - mean;
  red[tid] = d * d; __syncthreads();
  for (int s = 128; s > 0; s >>= 1) {
    if (tid < s) red[tid] += red[tid + s];
    __syncthreads();
  }
  float var = red[0] * (1.f / EE);
  out[base + tid] = d * rsqrtf(var + 1e-5f) * g[tid] + beta[tid];
}

// ---------------------------------------------------------------------------
// Deformable sampling: fuses P-softmax, offset->pixel, bilinear gather from
// f16 value map (b,pix,h,hd), weighted sum. One thread per (b,q,e) output.
// ---------------------------------------------------------------------------
__global__ __launch_bounds__(256)
void deform_kernel(const _Float16* __restrict__ valh,
                   const float* __restrict__ offv,
                   const float* __restrict__ awl,
                   const float* __restrict__ ref,
                   float* __restrict__ out) {
  int i = blockIdx.x * blockDim.x + threadIdx.x;   // over N*Q*E
  int e   = i & (EE - 1);
  int row = i >> 8;                                 // b*Q + q
  int b   = row >> 10;
  int h   = e >> 5;

  // softmax over P=4 attention logits
  const float* al = awl + (size_t)row * (NHH * PPTS) + h * PPTS;
  float a0 = al[0], a1 = al[1], a2 = al[2], a3 = al[3];
  float mx = fmaxf(fmaxf(a0, a1), fmaxf(a2, a3));
  float e0 = __expf(a0 - mx), e1 = __expf(a1 - mx);
  float e2 = __expf(a2 - mx), e3 = __expf(a3 - mx);
  float inv = 1.f / (e0 + e1 + e2 + e3);
  float wp[PPTS] = {e0 * inv, e1 * inv, e2 * inv, e3 * inv};

  float rx = ref[(size_t)row * 2 + 0];
  float ry = ref[(size_t)row * 2 + 1];
  const float* ofs = offv + (size_t)row * (NHH * PPTS * 2) + h * (PPTS * 2);

  float acc = 0.f;
#pragma unroll
  for (int p = 0; p < PPTS; ++p) {
    float x = rx * (float)WBEV + ofs[p * 2 + 0] - 0.5f;
    float y = ry * (float)HBEV + ofs[p * 2 + 1] - 0.5f;
    float x0f = floorf(x), y0f = floorf(y);
    float wx = x - x0f, wy = y - y0f;
    int x0 = (int)x0f, y0 = (int)y0f;
    float samp = 0.f;
#pragma unroll
    for (int dy = 0; dy < 2; ++dy) {
#pragma unroll
      for (int dx = 0; dx < 2; ++dx) {
        int xi = x0 + dx, yi = y0 + dy;
        bool valid = (xi >= 0) & (xi < WBEV) & (yi >= 0) & (yi < HBEV);
        int xc = min(max(xi, 0), WBEV - 1);
        int yc = min(max(yi, 0), HBEV - 1);
        float v = (float)valh[(size_t)(b * HWB + yc * WBEV + xc) * EE + e];
        float wgt = (dx ? wx : (1.f - wx)) * (dy ? wy : (1.f - wy));
        samp += valid ? wgt * v : 0.f;
      }
    }
    acc += wp[p] * samp;
  }
  out[(size_t)row * EE + e] = acc;
}

// ---------------------------------------------------------------------------
// Host orchestration
// ---------------------------------------------------------------------------
extern "C" void kernel_launch(void* const* d_in, const int* in_sizes, int n_in,
                              void* d_out, int out_size, void* d_ws, size_t ws_size,
                              hipStream_t stream) {
  const float* queries = (const float*)d_in[0];
  const float* bev     = (const float*)d_in[1];
  const float* ref     = (const float*)d_in[2];
  const float* Wq = (const float*)d_in[3],  * bq = (const float*)d_in[4];
  const float* Wk = (const float*)d_in[5],  * bk = (const float*)d_in[6];
  const float* Wv = (const float*)d_in[7],  * bv = (const float*)d_in[8];
  const float* Wo = (const float*)d_in[9],  * bo = (const float*)d_in[10];
  const float* ln1g = (const float*)d_in[11], * ln1b = (const float*)d_in[12];
  const float* Wval = (const float*)d_in[13], * bval = (const float*)d_in[14];
  const float* Woff = (const float*)d_in[15], * boff = (const float*)d_in[16];
  const float* Watn = (const float*)d_in[17], * batn = (const float*)d_in[18];
  const float* Wout = (const float*)d_in[19], * bout = (const float*)d_in[20];
  const float* ln2g = (const float*)d_in[21], * ln2b = (const float*)d_in[22];
  const float* W1 = (const float*)d_in[23], * b1 = (const float*)d_in[24];
  const float* W2 = (const float*)d_in[25], * b2 = (const float*)d_in[26];
  float* outp = (float*)d_out;

  // ---- workspace carve-out (~125 MB) ----
  size_t off = 0;
  auto take = [&](size_t bytes) -> void* {
    void* p = (char*)d_ws + off;
    off += (bytes + 255) & ~(size_t)255;
    return p;
  };
  _Float16*  hq   = (_Float16*)take((size_t)MROWS * EE * 2);
  _Float16*  hk   = (_Float16*)take((size_t)MROWS * EE * 2);
  float*     v32  = (float*)take((size_t)MROWS * EE * 4);
  _Float16*  vt   = (_Float16*)take((size_t)MROWS * EE * 2);
  float*     ctx  = (float*)take((size_t)MROWS * EE * 4);
  float*     o1   = (float*)take((size_t)MROWS * EE * 4);
  float*     out2 = (float*)take((size_t)MROWS * EE * 4);
  _Float16*  valh = (_Float16*)take((size_t)NB * HWB * EE * 2);
  float*     offv = (float*)take((size_t)MROWS * NHH * PPTS * 2 * 4);
  float*     awl  = (float*)take((size_t)MROWS * NHH * PPTS * 4);
  float*     dfo  = (float*)take((size_t)MROWS * EE * 4);
  float*     o3   = (float*)take((size_t)MROWS * EE * 4);
  float*     out4 = (float*)take((size_t)MROWS * EE * 4);
  float*     ffn  = (float*)take((size_t)MROWS * DFFN * 4);
  (void)ws_size; (void)n_in; (void)in_sizes; (void)out_size;

  auto gemm = [&](const float* A, const float* B, const float* bias,
                  float* Cf, _Float16* Ch, int M, int Nc, int K, int relu) {
    dim3 grid((Nc + 63) / 64, M / 128);
    gemm_wmma_kernel<<<grid, 256, 0, stream>>>(A, B, bias, Cf, Ch, M, Nc, K, relu);
  };
  const int ne = MROWS * EE;

  // ---- self attention ----
  gemm(queries, Wq, bq, nullptr, hq, MROWS, EE, EE, 0);   // f16 Q directly
  gemm(queries, Wk, bk, nullptr, hk, MROWS, EE, EE, 0);   // f16 K directly
  gemm(queries, Wv, bv, v32, nullptr, MROWS, EE, EE, 0);
  transpose_v_kernel<<<ne / 256, 256, 0, stream>>>(v32, vt);

  flash_attn_kernel<<<dim3(QN / 16 / 4, NHH, NB), 128, 0, stream>>>(hq, hk, vt, ctx);
  gemm(ctx, Wo, bo, o1, nullptr, MROWS, EE, EE, 0);
  ln_residual_kernel<<<MROWS, 256, 0, stream>>>(queries, o1, ln1g, ln1b, out2);

  // ---- deformable attention ----
  gemm(bev, Wval, bval, nullptr, valh, NB * HWB, EE, EE, 0);      // f16 value map
  gemm(out2, Woff, boff, offv, nullptr, MROWS, NHH * PPTS * 2, EE, 0);
  gemm(out2, Watn, batn, awl, nullptr, MROWS, NHH * PPTS, EE, 0);
  deform_kernel<<<ne / 256, 256, 0, stream>>>(valh, offv, awl, ref, dfo);
  gemm(dfo, Wout, bout, o3, nullptr, MROWS, EE, EE, 0);
  ln_residual_kernel<<<MROWS, 256, 0, stream>>>(out2, o3, ln2g, ln2b, out4);

  // ---- FFN ----
  gemm(out4, W1, b1, ffn, nullptr, MROWS, DFFN, EE, 1);           // fused ReLU
  gemm(ffn, W2, b2, outp, nullptr, MROWS, EE, DFFN, 0);
}
